// PraxisBlock_58128087384379
// MI455X (gfx1250) — matrix operand order repaired and checked by reading, not compiled, not run on hardware
//
#include <hip/hip_runtime.h>

#define Bsz 2
#define Ssz 2048
#define Dsz 768
#define Hsz 12
#define HDsz 64
#define Esz 8
#define Fsz 3072
#define Tsz 4096   // B*S

typedef __attribute__((ext_vector_type(16))) __bf16          v16bf;
typedef __attribute__((ext_vector_type(16))) unsigned short  v16u;
typedef __attribute__((ext_vector_type(8)))  float           v8f;

// ---------------- device helpers ----------------

static __device__ __forceinline__ v8f wmma_bf16(v16u a, v16u b, v8f c) {
  return __builtin_amdgcn_wmma_f32_16x16x32_bf16(
      false, __builtin_bit_cast(v16bf, a),
      false, __builtin_bit_cast(v16bf, b),
      (short)0, c, false, false);
}

static __device__ __forceinline__ unsigned short f2bf(float f) {
  unsigned u = __float_as_uint(f);
  unsigned r = u + 0x7FFFu + ((u >> 16) & 1u);   // round-to-nearest-even
  return (unsigned short)(r >> 16);
}

// A-fragment (16x32 bf16, MxK): lane = half*16 + m; element i -> K = (i&7) + ((i&8)<<1) + half*8
static __device__ __forceinline__ v16u load_a_pat(const unsigned short* p) {
  v16u a;
#pragma unroll
  for (int i = 0; i < 16; ++i) a[i] = p[(i & 7) + ((i & 8) << 1)];
  return a;
}

// B-fragment helper: element i -> K offset (i&7) + ((i&8)<<1) (caller adds half*8)
static __device__ __forceinline__ v16u load_b_contig(const unsigned short* p) {
  v16u a;
#pragma unroll
  for (int i = 0; i < 16; ++i) a[i] = p[(i & 7) + ((i & 8) << 1)];
  return a;
}

static __device__ __forceinline__ float gelu_tanh(float x) {
  float x3 = x * x * x;
  return 0.5f * x * (1.0f + tanhf(0.7978845608028654f * (x + 0.044715f * x3)));
}

// Stage a 32(K) x 64(N) f32 tile into LDS in B-FRAGMENT order: Bt[nt][lane][16].
// Fragment def: lane l (half=l>>4, ln=l&15), element i holds B[(i&7) + ((i&8)<<1) + half*8][nt*16 + ln].
// 256 threads: thread t writes 8 contiguous elements of one (nt,lane) as one uint4 LDS store.
static __device__ __forceinline__ void stage_b_tile(const float* __restrict__ Bw,
                                                    int kc, int n0, int Ndim,
                                                    unsigned short* __restrict__ Bt) {
  const int t  = threadIdx.x;
  const int nt = t >> 6;             // 0..3
  const int lf = (t >> 1) & 31;      // fragment lane
  const int ib = (t & 1) * 8;        // element base (0 or 8)
  const int kbase = (lf >> 4) * 8 + ib * 2;
  const int col = n0 + nt * 16 + (lf & 15);
  const float* src = Bw + (size_t)(kc + kbase) * Ndim + col;
  unsigned v[4];
#pragma unroll
  for (int jj = 0; jj < 4; ++jj) {
    unsigned lo = f2bf(src[(size_t)(2 * jj) * Ndim]);
    unsigned hi = f2bf(src[(size_t)(2 * jj + 1) * Ndim]);
    v[jj] = lo | (hi << 16);
  }
  ((uint4*)Bt)[t] = make_uint4(v[0], v[1], v[2], v[3]);
}

// Read one B fragment: 32 contiguous bytes per lane -> 2x ds_load_b128
static __device__ __forceinline__ v16u load_b_frag_lds(const unsigned short* __restrict__ Bt,
                                                       int nt, int lane) {
  return *(const v16u*)(Bt + ((size_t)nt * 32 + lane) * 16);
}

// ---------------- kernels ----------------

// h = rmsnorm(x, w) -> bf16
__global__ void rmsnorm_kernel(const float* __restrict__ x, const float* __restrict__ w,
                               unsigned short* __restrict__ out) {
  const int row = blockIdx.x;
  const float* xr = x + (size_t)row * Dsz;
  const int lane = threadIdx.x & 31, wvi = threadIdx.x >> 5;
  float ss = 0.f;
  for (int d = threadIdx.x; d < Dsz; d += 256) { float v = xr[d]; ss += v * v; }
  ss += __shfl_xor(ss, 16, 32); ss += __shfl_xor(ss, 8, 32);
  ss += __shfl_xor(ss, 4, 32);  ss += __shfl_xor(ss, 2, 32);
  ss += __shfl_xor(ss, 1, 32);
  __shared__ float red[8];
  if (lane == 0) red[wvi] = ss;
  __syncthreads();
  float tot = 0.f;
#pragma unroll
  for (int i = 0; i < 8; ++i) tot += red[i];
  const float rinv = rsqrtf(tot / (float)Dsz + 1e-6f);
  for (int d = threadIdx.x; d < Dsz; d += 256)
    out[(size_t)row * Dsz + d] = f2bf(xr[d] * rinv * w[d]);
}

// C[M,N] = A[M,K](bf16) @ B[K,N](f32 -> bf16 frag-order LDS, double-tile per barrier).
// MODE 0: store bf16. MODE 1: store f32 + residual.
// Block: 256 thr = 8 waves, tile 128(M) x 64(N); grid = (M/128, N/64). K,N compile-time.
template<int MODE, int ND, int KD>
__global__ void gemm_tpl(const unsigned short* __restrict__ A, const float* __restrict__ Bw,
                         const float* __restrict__ residual,
                         float* __restrict__ outf, unsigned short* __restrict__ outb) {
  __shared__ unsigned short Bt[2][2048];
  const int wvi = threadIdx.x >> 5, lane = threadIdx.x & 31;
  const int half = lane >> 4, ln = lane & 15;
  const int m0 = blockIdx.x * 128 + wvi * 16;
  const int n0 = blockIdx.y * 64;
  const unsigned short* Arow = A + (size_t)(m0 + ln) * KD;
  v8f acc[4] = {{}, {}, {}, {}};
  for (int kc = 0; kc < KD; kc += 64) {
    __syncthreads();
    stage_b_tile(Bw, kc,      n0, ND, Bt[0]);
    stage_b_tile(Bw, kc + 32, n0, ND, Bt[1]);
    __syncthreads();
    v16u a0 = load_a_pat(Arow + kc + half * 8);
    v16u a1 = load_a_pat(Arow + kc + 32 + half * 8);
#pragma unroll
    for (int nt = 0; nt < 4; ++nt)
      acc[nt] = wmma_bf16(a0, load_b_frag_lds(Bt[0], nt, lane), acc[nt]);
#pragma unroll
    for (int nt = 0; nt < 4; ++nt)
      acc[nt] = wmma_bf16(a1, load_b_frag_lds(Bt[1], nt, lane), acc[nt]);
  }
#pragma unroll
  for (int nt = 0; nt < 4; ++nt) {
#pragma unroll
    for (int r = 0; r < 8; ++r) {
      int row = m0 + r + 8 * half;
      int col = n0 + nt * 16 + ln;
      float v = acc[nt][r];
      if (MODE == 0) {
        outb[(size_t)row * ND + col] = f2bf(v);
      } else {
        outf[(size_t)row * ND + col] = v + residual[(size_t)row * ND + col];
      }
    }
  }
}

// Transpose vb [B,S,D](bf16) -> vt [B,D,S](bf16) with LDS tiling (coalesced both sides).
__global__ void transpose_v_kernel(const unsigned short* __restrict__ vb,
                                   unsigned short* __restrict__ vt) {
  __shared__ unsigned short tile[32][33];
  const int b  = blockIdx.z;
  const int s0 = blockIdx.x * 32;
  const int c0 = blockIdx.y * 32;
  const int tx = threadIdx.x & 31, ty = threadIdx.x >> 5;   // 32 x 8
#pragma unroll
  for (int k = 0; k < 4; ++k) {
    int sr = ty + k * 8;
    tile[sr][tx] = vb[((size_t)b * Ssz + s0 + sr) * Dsz + c0 + tx];
  }
  __syncthreads();
#pragma unroll
  for (int k = 0; k < 4; ++k) {
    int cr = ty + k * 8;
    vt[((size_t)b * Dsz + c0 + cr) * Ssz + s0 + tx] = tile[tx][cr];
  }
}

// Flash-style causal attention; one wave per 16-row query tile, 32 keys per step.
// Softmax is computed row-per-lane (in-lane, no cross-lane butterflies):
//   scores (C-layout) -> LDS S[16][32] f32 -> each lane reads its full row,
//   does max/exp/sum in registers, builds the P A-fragment directly.
// V is pre-transposed: VT[b, d, s] so PV B-fragments are two contiguous 16B loads per lane.
__global__ void attn_kernel(const unsigned short* __restrict__ Qb,
                            const unsigned short* __restrict__ Kb,
                            const unsigned short* __restrict__ VT,
                            const unsigned char* __restrict__ amask,
                            unsigned short* __restrict__ Ob) {
  __shared__ float Sf[4][16][32];   // per wave: raw masked scores, row-major
  __shared__ float Sr[4][16];       // per wave: per-row broadcast scratch
  const int wvi = threadIdx.x >> 5, lane = threadIdx.x & 31;
  const int half = lane >> 4, ln = lane & 15;
  const int bh = blockIdx.x, b = bh / Hsz, h = bh % Hsz;
  const int q0 = (blockIdx.y * 4 + wvi) * 16;
  const size_t basep = (size_t)b * Ssz * Dsz + (size_t)h * HDsz;
  const unsigned short* Q  = Qb + basep;
  const unsigned short* Kp = Kb + basep;
  const unsigned short* Vt0 = VT + ((size_t)b * Dsz + (size_t)h * HDsz + ln) * Ssz;
  float* Sw  = &Sf[wvi][0][0];
  float* Srw = &Sr[wvi][0];

  v16u aq0 = load_a_pat(Q + (size_t)(q0 + ln) * Dsz + half * 8);
  v16u aq1 = load_a_pat(Q + (size_t)(q0 + ln) * Dsz + 32 + half * 8);

  float m_run = -3.0e38f, l_run = 0.f;   // stats for row = ln (replicated in both halves)
  v8f o[4] = {{}, {}, {}, {}};

  for (int j0 = 0; j0 <= q0 + 15; j0 += 32) {
    // ---- scores (C-layout) -> LDS rows ----
#pragma unroll
    for (int t = 0; t < 2; ++t) {
      int k0 = j0 + 16 * t;
      if (k0 > q0 + 15) {              // only possible for t==1
#pragma unroll
        for (int r = 0; r < 8; ++r) Sw[(r + 8 * half) * 32 + 16 + ln] = -3.0e38f;
        continue;
      }
      int key = k0 + ln;
      int keyc = key < Ssz ? key : Ssz - 1;
      const unsigned short* kp = Kp + (size_t)keyc * Dsz + half * 8;
      v8f s = {};
      s = wmma_bf16(aq0, load_b_contig(kp), s);
      s = wmma_bf16(aq1, load_b_contig(kp + 32), s);
      bool kval = (key < Ssz) && (amask[(size_t)b * Ssz + key] != 0);
#pragma unroll
      for (int r = 0; r < 8; ++r) {
        int row = q0 + r + 8 * half;
        float vs = s[r] * 0.125f;                   // 1/sqrt(64)
        Sw[(r + 8 * half) * 32 + 16 * t + ln] = (kval && key <= row) ? vs : -3.0e38f;
      }
    }
    // ---- in-lane softmax over this lane's row (row = ln) ----
    float srow[32];
#pragma unroll
    for (int i = 0; i < 8; ++i)
      *(float4*)&srow[i * 4] = *(const float4*)&Sw[ln * 32 + i * 4];
    float mx = srow[0];
#pragma unroll
    for (int i = 1; i < 32; ++i) mx = fmaxf(mx, srow[i]);
    float mnew = fmaxf(m_run, mx);
    float sc = (m_run <= -1.0e38f) ? 0.f : __expf(m_run - mnew);
    float p[32]; float sum = 0.f;
#pragma unroll
    for (int i = 0; i < 32; ++i) {
      p[i] = (srow[i] <= -1.0e38f) ? 0.f : __expf(srow[i] - mnew);
      sum += p[i];
    }
    l_run = l_run * sc + sum;
    m_run = mnew;
    // ---- broadcast per-row rescale to C-layout O accumulators ----
    if (half == 0) Srw[ln] = sc;
    float scr[8];
#pragma unroll
    for (int r = 0; r < 8; ++r) scr[r] = Srw[r + 8 * half];
#pragma unroll
    for (int t2 = 0; t2 < 4; ++t2)
#pragma unroll
      for (int r = 0; r < 8; ++r) o[t2][r] *= scr[r];
    // ---- P A-fragment directly from registers (compile-time picks) ----
    v16u ap;
#pragma unroll
    for (int i = 0; i < 16; ++i)
      ap[i] = f2bf(p[(i & 7) + ((i & 8) << 1) + half * 8]);
    // ---- PV ----
#pragma unroll
    for (int t2 = 0; t2 < 4; ++t2) {
      v16u bv = load_b_contig(Vt0 + (size_t)t2 * 16 * Ssz + j0 + half * 8);
      o[t2] = wmma_bf16(ap, bv, o[t2]);
    }
  }
  // ---- final 1/l normalization (broadcast via LDS) ----
  if (half == 0) Srw[ln] = l_run;
  float lr[8];
#pragma unroll
  for (int r = 0; r < 8; ++r) lr[r] = Srw[r + 8 * half];
#pragma unroll
  for (int t2 = 0; t2 < 4; ++t2) {
#pragma unroll
    for (int r = 0; r < 8; ++r) {
      int row = q0 + r + 8 * half;
      Ob[((size_t)b * Ssz + row) * Dsz + (size_t)h * HDsz + t2 * 16 + ln] =
          f2bf(o[t2][r] / lr[r]);
    }
  }
}

__global__ void zero_kernel(int* cnt, float* ps) {
  int i = threadIdx.x;
  if (i < Esz) { cnt[i] = 0; ps[i] = 0.f; }
}

// Per-token router: rmsnorm + logits + softmax + top-2; build per-expert token lists.
__global__ void route_kernel(const float* __restrict__ x1, const float* __restrict__ nw,
                             const float* __restrict__ rw, const float* __restrict__ rb,
                             int* __restrict__ cnt, float* __restrict__ probsum,
                             int* __restrict__ ltok, int* __restrict__ lk,
                             float* __restrict__ tscore) {
  int tok = blockIdx.x * 256 + threadIdx.x;
  if (tok >= Tsz) return;
  const float* xr = x1 + (size_t)tok * Dsz;
  float ss = 0.f;
  for (int d = 0; d < Dsz; ++d) { float v = xr[d]; ss += v * v; }
  float rinv = rsqrtf(ss / (float)Dsz + 1e-6f);
  float lg[Esz];
#pragma unroll
  for (int e = 0; e < Esz; ++e) lg[e] = rb[e];
  for (int d = 0; d < Dsz; ++d) {
    float xn = xr[d] * rinv * nw[d];
#pragma unroll
    for (int e = 0; e < Esz; ++e) lg[e] += xn * rw[d * Esz + e];
  }
  float mx = lg[0];
#pragma unroll
  for (int e = 1; e < Esz; ++e) mx = fmaxf(mx, lg[e]);
  float pe[Esz]; float sum = 0.f;
#pragma unroll
  for (int e = 0; e < Esz; ++e) { pe[e] = __expf(lg[e] - mx); sum += pe[e]; }
  float inv = 1.f / sum;
#pragma unroll
  for (int e = 0; e < Esz; ++e) { pe[e] *= inv; atomicAdd(&probsum[e], pe[e]); }
  int b0 = 0; float v0 = pe[0];
#pragma unroll
  for (int e = 1; e < Esz; ++e) if (pe[e] > v0) { v0 = pe[e]; b0 = e; }
  int b1 = -1; float v1 = -1.f;
#pragma unroll
  for (int e = 0; e < Esz; ++e) if (e != b0 && pe[e] > v1) { v1 = pe[e]; b1 = e; }
  tscore[tok * 2 + 0] = v0;
  tscore[tok * 2 + 1] = v1;
  int s0 = atomicAdd(&cnt[b0], 1); ltok[b0 * Tsz + s0] = tok; lk[b0 * Tsz + s0] = 0;
  int s1 = atomicAdd(&cnt[b1], 1); ltok[b1 * Tsz + s1] = tok; lk[b1 * Tsz + s1] = 1;
}

__global__ void finalize_kernel(const int* __restrict__ cnt, const float* __restrict__ probsum,
                                int* __restrict__ basev, float* __restrict__ tail) {
  if (threadIdx.x == 0) {
    int b = 0; float loss = 0.f;
    for (int e = 0; e < Esz; ++e) {
      basev[e] = b; b += cnt[e];
      float me = probsum[e] / (float)Tsz;
      float fe = (float)cnt[e] / (float)Tsz;
      loss += me * fe;
      tail[1 + e] = (float)cnt[e];
    }
    tail[0] = (float)Esz * loss;
  }
}

// Grouped expert GEMM1: hid = gelu(gather(h2b) @ w1[e] + b1[e]) -> bf16
__global__ void egemm1_kernel(const unsigned short* __restrict__ h2b,
                              const float* __restrict__ w1, const float* __restrict__ b1,
                              const int* __restrict__ cnt, const int* __restrict__ basev,
                              const int* __restrict__ ltok,
                              unsigned short* __restrict__ hidb) {
  const int e = blockIdx.z;
  const int c = cnt[e];
  if ((int)(blockIdx.y * 128) >= c) return;
  __shared__ unsigned short Bt[2][2048];
  const int wvi = threadIdx.x >> 5, lane = threadIdx.x & 31;
  const int half = lane >> 4, ln = lane & 15;
  const int m0 = blockIdx.y * 128 + wvi * 16;
  const int n0 = blockIdx.x * 64;
  const float* Bw = w1 + (size_t)e * Dsz * Fsz;
  int m = m0 + ln;
  int slot = m < c ? m : c - 1;
  int tok = ltok[e * Tsz + slot];
  const unsigned short* Arow = h2b + (size_t)tok * Dsz;
  v8f acc[4] = {{}, {}, {}, {}};
  for (int kc = 0; kc < Dsz; kc += 64) {
    __syncthreads();
    stage_b_tile(Bw, kc,      n0, Fsz, Bt[0]);
    stage_b_tile(Bw, kc + 32, n0, Fsz, Bt[1]);
    __syncthreads();
    v16u a0 = load_a_pat(Arow + kc + half * 8);
    v16u a1 = load_a_pat(Arow + kc + 32 + half * 8);
#pragma unroll
    for (int nt = 0; nt < 4; ++nt)
      acc[nt] = wmma_bf16(a0, load_b_frag_lds(Bt[0], nt, lane), acc[nt]);
#pragma unroll
    for (int nt = 0; nt < 4; ++nt)
      acc[nt] = wmma_bf16(a1, load_b_frag_lds(Bt[1], nt, lane), acc[nt]);
  }
  const int ob = basev[e];
#pragma unroll
  for (int nt = 0; nt < 4; ++nt) {
#pragma unroll
    for (int r = 0; r < 8; ++r) {
      int rm = m0 + r + 8 * half;
      if (rm < c) {
        int col = n0 + nt * 16 + ln;
        float v = acc[nt][r] + b1[(size_t)e * Fsz + col];
        hidb[(size_t)(ob + rm) * Fsz + col] = f2bf(gelu_tanh(v));
      }
    }
  }
}

// Grouped expert GEMM2: eo[tok][kslot] = hid @ w2[e] + b2[e]  (unweighted)
__global__ void egemm2_kernel(const unsigned short* __restrict__ hidb,
                              const float* __restrict__ w2, const float* __restrict__ b2,
                              const int* __restrict__ cnt, const int* __restrict__ basev,
                              const int* __restrict__ ltok, const int* __restrict__ lk,
                              float* __restrict__ eo) {
  const int e = blockIdx.z;
  const int c = cnt[e];
  if ((int)(blockIdx.y * 128) >= c) return;
  __shared__ unsigned short Bt[2][2048];
  const int wvi = threadIdx.x >> 5, lane = threadIdx.x & 31;
  const int half = lane >> 4, ln = lane & 15;
  const int m0 = blockIdx.y * 128 + wvi * 16;
  const int n0 = blockIdx.x * 64;
  const float* Bw = w2 + (size_t)e * Fsz * Dsz;
  int m = m0 + ln;
  int slot = m < c ? m : c - 1;
  const unsigned short* Arow = hidb + (size_t)(basev[e] + slot) * Fsz;
  v8f acc[4] = {{}, {}, {}, {}};
  for (int kc = 0; kc < Fsz; kc += 64) {
    __syncthreads();
    stage_b_tile(Bw, kc,      n0, Dsz, Bt[0]);
    stage_b_tile(Bw, kc + 32, n0, Dsz, Bt[1]);
    __syncthreads();
    v16u a0 = load_a_pat(Arow + kc + half * 8);
    v16u a1 = load_a_pat(Arow + kc + 32 + half * 8);
#pragma unroll
    for (int nt = 0; nt < 4; ++nt)
      acc[nt] = wmma_bf16(a0, load_b_frag_lds(Bt[0], nt, lane), acc[nt]);
#pragma unroll
    for (int nt = 0; nt < 4; ++nt)
      acc[nt] = wmma_bf16(a1, load_b_frag_lds(Bt[1], nt, lane), acc[nt]);
  }
#pragma unroll
  for (int nt = 0; nt < 4; ++nt) {
#pragma unroll
    for (int r = 0; r < 8; ++r) {
      int rm = m0 + r + 8 * half;
      if (rm < c) {
        int col = n0 + nt * 16 + ln;
        int tok = ltok[e * Tsz + rm];
        int ks  = lk[e * Tsz + rm];
        eo[((size_t)tok * 2 + ks) * Dsz + col] = acc[nt][r] + b2[(size_t)e * Dsz + col];
      }
    }
  }
}

// Faithful to the reference's raw reshape([T,K,D] -> [K,B,S,D]) weighted combine.
__global__ void combine_kernel(float* __restrict__ out, const float* __restrict__ eo,
                               const float* __restrict__ tscore) {
  size_t i = (size_t)blockIdx.x * 256 + threadIdx.x;
  size_t t2 = i / Dsz;
  int d = (int)(i % Dsz);
  size_t ta = t2 >> 1;                    int ka = (int)(t2 & 1);
  size_t tb = (Tsz >> 1) + (t2 >> 1);     int kb = (int)(t2 & 1);
  float s0 = tscore[t2 * 2 + 0], s1 = tscore[t2 * 2 + 1];
  out[i] += eo[(ta * 2 + ka) * Dsz + d] * s0 + eo[(tb * 2 + kb) * Dsz + d] * s1;
}

// ---------------- host ----------------

static size_t carve_off(size_t& off, size_t bytes) {
  size_t r = off;
  off += (bytes + 255) & ~(size_t)255;
  return r;
}

extern "C" void kernel_launch(void* const* d_in, const int* in_sizes, int n_in,
                              void* d_out, int out_size, void* d_ws, size_t ws_size,
                              hipStream_t stream) {
  (void)in_sizes; (void)n_in; (void)out_size; (void)ws_size;
  const float* x            = (const float*)d_in[0];
  const unsigned char* amsk = (const unsigned char*)d_in[1];
  const float* anw = (const float*)d_in[2];
  const float* wq  = (const float*)d_in[3];
  const float* wk  = (const float*)d_in[4];
  const float* wvp = (const float*)d_in[5];
  const float* wo  = (const float*)d_in[6];
  const float* mnw = (const float*)d_in[7];
  const float* rw  = (const float*)d_in[8];
  const float* rb  = (const float*)d_in[9];
  const float* w1  = (const float*)d_in[10];
  const float* b1  = (const float*)d_in[11];
  const float* w2  = (const float*)d_in[12];
  const float* b2  = (const float*)d_in[13];
  float* out = (float*)d_out;

  char* base = (char*)d_ws;
  size_t off = 0;
  const size_t TD = (size_t)Tsz * Dsz;
  unsigned short* hb   = (unsigned short*)(base + carve_off(off, TD * 2));
  unsigned short* qb   = (unsigned short*)(base + carve_off(off, TD * 2));
  unsigned short* kb   = (unsigned short*)(base + carve_off(off, TD * 2));
  unsigned short* vb   = (unsigned short*)(base + carve_off(off, TD * 2));
  unsigned short* vt   = (unsigned short*)(base + carve_off(off, TD * 2 + 256)); // +tail pad
  unsigned short* aob  = (unsigned short*)(base + carve_off(off, TD * 2));
  unsigned short* h2b  = (unsigned short*)(base + carve_off(off, TD * 2));
  unsigned short* hidb = (unsigned short*)(base + carve_off(off, (size_t)2 * Tsz * Fsz * 2));
  float* eo     = (float*)(base + carve_off(off, (size_t)Tsz * 2 * Dsz * 4));
  float* tscore = (float*)(base + carve_off(off, (size_t)Tsz * 2 * 4));
  int*   ltok   = (int*)(base + carve_off(off, (size_t)Esz * Tsz * 4));
  int*   lkk    = (int*)(base + carve_off(off, (size_t)Esz * Tsz * 4));
  int*   cnt    = (int*)(base + carve_off(off, 64));
  float* probsum= (float*)(base + carve_off(off, 64));
  int*   basev  = (int*)(base + carve_off(off, 64));

  // ---- attention block ----
  rmsnorm_kernel<<<Tsz, 256, 0, stream>>>(x, anw, hb);
  dim3 g768(Tsz / 128, Dsz / 64);
  gemm_tpl<0, Dsz, Dsz><<<g768, 256, 0, stream>>>(hb, wq,  nullptr, nullptr, qb);
  gemm_tpl<0, Dsz, Dsz><<<g768, 256, 0, stream>>>(hb, wk,  nullptr, nullptr, kb);
  gemm_tpl<0, Dsz, Dsz><<<g768, 256, 0, stream>>>(hb, wvp, nullptr, nullptr, vb);
  transpose_v_kernel<<<dim3(Ssz / 32, Dsz / 32, Bsz), 256, 0, stream>>>(vb, vt);
  attn_kernel<<<dim3(Bsz * Hsz, Ssz / 64), 128, 0, stream>>>(qb, kb, vt, amsk, aob);
  gemm_tpl<1, Dsz, Dsz><<<g768, 256, 0, stream>>>(aob, wo, x, out, nullptr);   // x1 -> d_out

  // ---- MoE block ----
  rmsnorm_kernel<<<Tsz, 256, 0, stream>>>(out, mnw, h2b);
  zero_kernel<<<1, 32, 0, stream>>>(cnt, probsum);
  route_kernel<<<Tsz / 256, 256, 0, stream>>>(out, mnw, rw, rb, cnt, probsum, ltok, lkk, tscore);
  finalize_kernel<<<1, 1, 0, stream>>>(cnt, probsum, basev, out + TD);
  egemm1_kernel<<<dim3(Fsz / 64, Tsz / 128, Esz), 256, 0, stream>>>(h2b, w1, b1, cnt, basev, ltok, hidb);
  egemm2_kernel<<<dim3(Dsz / 64, Tsz / 128, Esz), 256, 0, stream>>>(hidb, w2, b2, cnt, basev, ltok, lkk, eo);
  combine_kernel<<<(unsigned)(TD / 256), 256, 0, stream>>>(out, eo, tscore);
}
